// BotRGCN_13056700580140
// MI455X (gfx1250) — matrix operand order repaired and checked by reading, not compiled, not run on hardware
//
#include <hip/hip_runtime.h>

// ---------------------------------------------------------------------------
// Types for CDNA5 WMMA
// ---------------------------------------------------------------------------
typedef __bf16 bf16x16 __attribute__((ext_vector_type(16)));
typedef float  v8f     __attribute__((ext_vector_type(8)));

union FragAB { bf16x16 v; uint4 q[2]; };
union FragC  { v8f v; float f[8]; };

// Packed f32 -> bf16 conversion through the HW converter (v_cvt_pk_bf16_f32).
__device__ __forceinline__ unsigned int pkbf(float a, float b) {
    union { __bf16 h[2]; unsigned int u; } x;
    x.h[0] = (__bf16)a;
    x.h[1] = (__bf16)b;
    return x.u;
}

// ---------------------------------------------------------------------------
// Tiled GEMM: C[N, nblk*128] = act(A[N,K] @ B[K,128-per-block] + bias)
// BM=128, BN=128, BK=32; 256 threads = 8 wave32; wave w owns rows w*16..w*16+15.
// A staged in LDS as bf16 [m][k] (row-major), B staged as bf16 [n][k]
// (column-major) so WMMA fragments are contiguous 16B LDS loads.
// ---------------------------------------------------------------------------
__global__ __launch_bounds__(256) void gemm_bf16_wmma(
    const float* __restrict__ A, int N, int K, int lda,
    const float* __restrict__ B0, const float* __restrict__ B1,
    const float* __restrict__ B2,
    const float* __restrict__ bias, float* __restrict__ C, int ldc, int leaky)
{
    __shared__ unsigned short lA[128 * 32];   // [m][k]  8 KB
    __shared__ unsigned short lB[128 * 32];   // [n][k]  8 KB

    const int t    = threadIdx.x;
    const int row0 = blockIdx.x * 128;
    const int cb   = blockIdx.y;              // column block (0..nblk-1)
    const float* Bsel = (cb == 0) ? B0 : (cb == 1) ? B1 : B2;
    const int colOff = cb * 128;

    const int w    = t >> 5;                  // wave id 0..7
    const int lane = t & 31;
    const int half = lane >> 4;
    const int mr   = lane & 15;

    v8f acc[8];
#pragma unroll
    for (int g = 0; g < 8; ++g) acc[g] = (v8f)0.0f;

    const uint4* lA4 = (const uint4*)lA;
    const uint4* lB4 = (const uint4*)lB;

    const int kSteps = K >> 5;
    for (int kt = 0; kt < kSteps; ++kt) {
        const int k0 = kt << 5;

        // ---- stage A tile (128 rows x 32 k), fp32 -> bf16, [m][k] ----
#pragma unroll
        for (int i = 0; i < 4; ++i) {
            int s   = t + i * 256;            // 0..1023 quads
            int row = s >> 3;                 // 0..127
            int kq  = (s & 7) << 2;           // 0,4,..,28
            float4 v = make_float4(0.f, 0.f, 0.f, 0.f);
            int grow = row0 + row;
            if (grow < N)
                v = *(const float4*)(A + (size_t)grow * lda + k0 + kq);
            uint2 p;
            p.x = pkbf(v.x, v.y);
            p.y = pkbf(v.z, v.w);
            *(uint2*)&lA[row * 32 + kq] = p;
        }
        // ---- stage B tile (32 k x 128 n) into [n][k] ----
        // Each thread: 4-deep K column chunk at fixed n. The 4 global b32
        // loads are lane-coalesced (consecutive lanes -> consecutive n);
        // pack 4 bf16 -> single ds_store_b64 (no 16-bit scatter stores).
#pragma unroll
        for (int i = 0; i < 4; ++i) {
            int s  = t + i * 256;             // 0..1023 = 128 n x 8 kq
            int n  = s & 127;
            int kq = (s >> 7) << 2;           // 0,4,..,28
            const float* bp = Bsel + (size_t)(k0 + kq) * 128 + n;
            float v0 = bp[0];
            float v1 = bp[128];
            float v2 = bp[256];
            float v3 = bp[384];
            uint2 p;
            p.x = pkbf(v0, v1);
            p.y = pkbf(v2, v3);
            *(uint2*)&lB[n * 32 + kq] = p;
        }
        __syncthreads();

        // ---- A fragment for this wave's 16-row strip ----
        FragAB fa;
        {
            int row = w * 16 + mr;
            fa.q[0] = lA4[row * 4 + half];        // K = half*8 .. half*8+7
            fa.q[1] = lA4[row * 4 + 2 + half];    // K = 16+half*8 .. +7
        }
        // ---- prefetch all 8 B fragments, then dense WMMA chain ----
        FragAB fb[8];
#pragma unroll
        for (int g = 0; g < 8; ++g) {
            int n = g * 16 + mr;
            fb[g].q[0] = lB4[n * 4 + half];          // K = half*8 .. half*8+7
            fb[g].q[1] = lB4[n * 4 + 2 + half];      // K = 16+half*8 .. +7
        }
#pragma unroll
        for (int g = 0; g < 8; ++g) {
            acc[g] = __builtin_amdgcn_wmma_f32_16x16x32_bf16(
                false, fa.v, false, fb[g].v, (short)0, acc[g], false, false);
        }
        __syncthreads();
    }

    // ---- epilogue: bias + leaky, store per C layout ----
#pragma unroll
    for (int g = 0; g < 8; ++g) {
        int colInBlk = g * 16 + mr;
        float bv = bias ? bias[colInBlk] : 0.0f;
        FragC fc; fc.v = acc[g];
#pragma unroll
        for (int r = 0; r < 8; ++r) {
            int grow = row0 + w * 16 + half * 8 + r;
            if (grow < N) {
                float v = fc.f[r] + bv;
                if (leaky) v = (v > 0.0f) ? v : 0.01f * v;
                C[(size_t)grow * ldc + colOff + colInBlk] = v;
            }
        }
    }
}

// ---------------------------------------------------------------------------
// RGCN aggregation kernels
// ---------------------------------------------------------------------------
__global__ void count_edges(const int* __restrict__ dst,
                            const int* __restrict__ et, int E,
                            float* __restrict__ cnt)
{
    int e = blockIdx.x * blockDim.x + threadIdx.x;
    if (e < E) unsafeAtomicAdd(&cnt[(size_t)dst[e] * 2 + et[e]], 1.0f);
}

__global__ void scatter_edges(const float* __restrict__ cat,
                              const int* __restrict__ src,
                              const int* __restrict__ dst,
                              const int* __restrict__ et, int E,
                              float* __restrict__ sums)
{
    long long gid = (long long)blockIdx.x * blockDim.x + threadIdx.x;
    int e = (int)(gid >> 5);
    if (e >= E) return;
    int part = (int)(gid & 31) << 2;          // 4 floats per thread
    int s = src[e], d = dst[e], r = et[e];
    float4 v = *(const float4*)(cat + (size_t)s * 384 + r * 128 + part);
    float* p = sums + ((size_t)d * 2 + r) * 128 + part;
    unsafeAtomicAdd(p + 0, v.x);
    unsafeAtomicAdd(p + 1, v.y);
    unsafeAtomicAdd(p + 2, v.z);
    unsafeAtomicAdd(p + 3, v.w);
}

__global__ void combine_rgcn(const float* __restrict__ sums,
                             const float* __restrict__ cnt,
                             const float* __restrict__ cat,
                             const float* __restrict__ brgcn,
                             float* __restrict__ out, int N)
{
    long long gid = (long long)blockIdx.x * blockDim.x + threadIdx.x;
    int n = (int)(gid >> 5);
    if (n >= N) return;
    int part = (int)(gid & 31) << 2;
    float c0 = fmaxf(cnt[(size_t)n * 2 + 0], 1.0f);
    float c1 = fmaxf(cnt[(size_t)n * 2 + 1], 1.0f);
    float4 s0 = *(const float4*)(sums + ((size_t)n * 2 + 0) * 128 + part);
    float4 s1 = *(const float4*)(sums + ((size_t)n * 2 + 1) * 128 + part);
    float4 rt = *(const float4*)(cat + (size_t)n * 384 + 256 + part);
    float4 b  = *(const float4*)(brgcn + part);
    float4 o;
    o.x = s0.x / c0 + s1.x / c1 + rt.x + b.x;
    o.y = s0.y / c0 + s1.y / c1 + rt.y + b.y;
    o.z = s0.z / c0 + s1.z / c1 + rt.z + b.z;
    o.w = s0.w / c0 + s1.w / c1 + rt.w + b.w;
    *(float4*)(out + (size_t)n * 128 + part) = o;
}

// ---------------------------------------------------------------------------
// Final head: out[N,2] = x[N,128] @ Wout[128,2] + bout
// ---------------------------------------------------------------------------
__global__ __launch_bounds__(256) void head_gemv(
    const float* __restrict__ x, const float* __restrict__ Wout,
    const float* __restrict__ bout, float* __restrict__ out, int N)
{
    __shared__ float w[256];
    w[threadIdx.x] = Wout[threadIdx.x];
    __syncthreads();
    int n = blockIdx.x * blockDim.x + threadIdx.x;
    if (n >= N) return;
    float a0 = bout[0], a1 = bout[1];
    const float4* row = (const float4*)(x + (size_t)n * 128);
#pragma unroll
    for (int j = 0; j < 32; ++j) {
        float4 v = row[j];
        int c = j * 4;
        a0 += v.x * w[(c + 0) * 2] + v.y * w[(c + 1) * 2] +
              v.z * w[(c + 2) * 2] + v.w * w[(c + 3) * 2];
        a1 += v.x * w[(c + 0) * 2 + 1] + v.y * w[(c + 1) * 2 + 1] +
              v.z * w[(c + 2) * 2 + 1] + v.w * w[(c + 3) * 2 + 1];
    }
    out[(size_t)n * 2 + 0] = a0;
    out[(size_t)n * 2 + 1] = a1;
}

// ---------------------------------------------------------------------------
// Host launcher
// ---------------------------------------------------------------------------
static inline int cdiv(long long a, long long b) { return (int)((a + b - 1) / b); }

extern "C" void kernel_launch(void* const* d_in, const int* in_sizes, int n_in,
                              void* d_out, int out_size, void* d_ws, size_t ws_size,
                              hipStream_t stream)
{
    (void)n_in; (void)out_size; (void)ws_size;
    const float* tweet  = (const float*)d_in[0];
    const int*   eidx   = (const int*)d_in[1];
    const int*   etype  = (const int*)d_in[2];
    const float* Wt     = (const float*)d_in[3];
    const float* bt     = (const float*)d_in[4];
    const float* Wi     = (const float*)d_in[5];
    const float* bi     = (const float*)d_in[6];
    const float* Wrel   = (const float*)d_in[7];   // [2,128,128]
    const float* Wroot  = (const float*)d_in[8];
    const float* brgcn  = (const float*)d_in[9];
    const float* Wo     = (const float*)d_in[10];
    const float* bo     = (const float*)d_in[11];
    const float* Wout   = (const float*)d_in[12];
    const float* bout   = (const float*)d_in[13];

    const int T = in_sizes[3] / 128;               // 768
    const int N = in_sizes[0] / T;                 // 100000
    const int E = in_sizes[2];                     // 600000
    const int* src = eidx;
    const int* dst = eidx + E;

    // workspace layout (floats): requires N*898*4 bytes (~360 MB)
    float* ws   = (float*)d_ws;
    float* bx   = ws;                              // [N,128]
    float* by   = bx + (size_t)N * 128;            // [N,128]
    float* bcat = by + (size_t)N * 128;            // [N,384] = xr0|xr1|xroot
    float* sums = bcat + (size_t)N * 384;          // [N,2,128]
    float* cnt  = sums + (size_t)N * 256;          // [N,2]

    const dim3 blk(256);
    const int gN = cdiv(N, 128);

    // edge counts (same for both RGCN layers)
    hipMemsetAsync(cnt, 0, (size_t)N * 2 * sizeof(float), stream);
    count_edges<<<cdiv(E, 256), blk, 0, stream>>>(dst, etype, E, cnt);

    // x = leaky(tweet @ Wt + bt)
    gemm_bf16_wmma<<<dim3(gN, 1), blk, 0, stream>>>(
        tweet, N, T, T, Wt, Wt, Wt, bt, bx, 128, 1);
    // x = leaky(x @ Wi + bi)
    gemm_bf16_wmma<<<dim3(gN, 1), blk, 0, stream>>>(
        bx, N, 128, 128, Wi, Wi, Wi, bi, by, 128, 1);

    // ---- RGCN layer 1 : in by -> out bx ----
    gemm_bf16_wmma<<<dim3(gN, 3), blk, 0, stream>>>(
        by, N, 128, 128, Wrel, Wrel + 128 * 128, Wroot, nullptr, bcat, 384, 0);
    hipMemsetAsync(sums, 0, (size_t)N * 256 * sizeof(float), stream);
    scatter_edges<<<cdiv((long long)E * 32, 256), blk, 0, stream>>>(
        bcat, src, dst, etype, E, sums);
    combine_rgcn<<<cdiv((long long)N * 32, 256), blk, 0, stream>>>(
        sums, cnt, bcat, brgcn, bx, N);

    // ---- RGCN layer 2 : in bx -> out by ----
    gemm_bf16_wmma<<<dim3(gN, 3), blk, 0, stream>>>(
        bx, N, 128, 128, Wrel, Wrel + 128 * 128, Wroot, nullptr, bcat, 384, 0);
    hipMemsetAsync(sums, 0, (size_t)N * 256 * sizeof(float), stream);
    scatter_edges<<<cdiv((long long)E * 32, 256), blk, 0, stream>>>(
        bcat, src, dst, etype, E, sums);
    combine_rgcn<<<cdiv((long long)N * 32, 256), blk, 0, stream>>>(
        sums, cnt, bcat, brgcn, by, N);

    // x = leaky(x @ Wo + bo)
    gemm_bf16_wmma<<<dim3(gN, 1), blk, 0, stream>>>(
        by, N, 128, 128, Wo, Wo, Wo, bo, bx, 128, 1);
    // out = x @ Wout + bout
    head_gemv<<<cdiv(N, 256), blk, 0, stream>>>(
        bx, Wout, bout, (float*)d_out, N);
}
